// Boundaryloss_73469710565586
// MI455X (gfx1250) — matrix operand order repaired and checked by reading, not compiled, not run on hardware
//
#include <hip/hip_runtime.h>
#include <hip/hip_bf16.h>

typedef __attribute__((ext_vector_type(16))) _Float16 v16h;
typedef __attribute__((ext_vector_type(8)))  _Float16 v8h;
typedef __attribute__((ext_vector_type(8)))  float    v8f;

#define NB    16                  // batch
#define NC    21                  // classes
#define HH    41                  // prob map h=w
#define NPIX  (HH*HH)             // 1681
#define NP    1696                // pixels padded to 53*32
#define CP    32                  // channels padded
#define IMH   321                 // image h=w
#define KSTEPS (NP/32)            // 53
#define EPSV  1e-5f
#define COMPAT_G 3.0f
#define COMPAT_B 10.0f
#define INV_2SG2 8.0f             // 1/(2*(3/12)^2)
#define INV_2SB2 0.01125f         // 1/(2*(80/12)^2) = 144/12800
#define INV_2SR2 (1.0f/338.0f)    // 1/(2*13^2)

__global__ void zero_kernel(float* out) { out[0] = 0.0f; }

// probs = renorm(clamp(softmax(predict), eps, 1)); logu = log(probs)
__global__ void probs_kernel(const float* __restrict__ predict,
                             float* __restrict__ probs, float* __restrict__ logu) {
  int idx = blockIdx.x * blockDim.x + threadIdx.x;
  if (idx >= NB * NPIX) return;
  int n = idx / NPIX, i = idx % NPIX;
  const float* p = predict + (size_t)n * NC * NPIX + i;
  float m = -1e30f;
  float e[NC];
#pragma unroll
  for (int c = 0; c < NC; ++c) m = fmaxf(m, p[(size_t)c * NPIX]);
  float s = 0.f;
#pragma unroll
  for (int c = 0; c < NC; ++c) { e[c] = __expf(p[(size_t)c * NPIX] - m); s += e[c]; }
  float inv = 1.0f / s, s2 = 0.f;
#pragma unroll
  for (int c = 0; c < NC; ++c) { e[c] = fminf(fmaxf(e[c] * inv, EPSV), 1.0f); s2 += e[c]; }
  float inv2 = 1.0f / s2;
  float* po = probs + (size_t)n * NC * NPIX + i;
  float* lo = logu  + (size_t)n * NC * NPIX + i;
#pragma unroll
  for (int c = 0; c < NC; ++c) {
    float q = e[c] * inv2;
    po[(size_t)c * NPIX] = q;
    lo[(size_t)c * NPIX] = __logf(q);
  }
}

// antialiased linear resize (jax.image.resize 'linear' with antialias) 321->41, NHWC gather
__global__ void resize_kernel(const float* __restrict__ images, float* __restrict__ colors) {
  int idx = blockIdx.x * blockDim.x + threadIdx.x;
  if (idx >= NB * NPIX) return;
  int n = idx / NPIX, i = idx % NPIX;
  int oy = i / HH, ox = i % HH;
  const float s = (float)IMH / (float)HH;     // ~7.829 (downscale -> triangle radius s)
  float cy = (oy + 0.5f) * s - 0.5f;
  float cx = (ox + 0.5f) * s - 0.5f;
  int y0 = (int)ceilf(cy - s), y1 = (int)floorf(cy + s);
  int x0 = (int)ceilf(cx - s), x1 = (int)floorf(cx + s);
  float wxs = 0.f, wys = 0.f;
  for (int x = x0; x <= x1; ++x)
    if (x >= 0 && x < IMH) wxs += 1.0f - fabsf((float)x - cx) / s;
  float a0 = 0.f, a1 = 0.f, a2 = 0.f;
  const float* base = images + (size_t)n * 3 * IMH * IMH;
  for (int y = y0; y <= y1; ++y) {
    if (y < 0 || y >= IMH) continue;
    float wy = 1.0f - fabsf((float)y - cy) / s;
    wys += wy;
    for (int x = x0; x <= x1; ++x) {
      if (x < 0 || x >= IMH) continue;
      float w = wy * (1.0f - fabsf((float)x - cx) / s);
      size_t o = (size_t)y * IMH + x;
      a0 += w * base[o];
      a1 += w * base[(size_t)IMH * IMH + o];
      a2 += w * base[2 * (size_t)IMH * IMH + o];
    }
  }
  float norm = 1.0f / (wys * wxs);
  float* cp = colors + ((size_t)n * NPIX + i) * 3;
  cp[0] = a0 * norm; cp[1] = a1 * norm; cp[2] = a2 * norm;
}

// row sums of spatial Gaussian kernel (shared over batch): one wave per row
__global__ void sg_kernel(float* __restrict__ sg) {
  int wid  = (blockIdx.x * blockDim.x + threadIdx.x) >> 5;
  int lane = threadIdx.x & 31;
  if (wid >= NPIX) return;
  int yi = wid / HH, xi = wid % HH;
  float s = 0.f;
  for (int j = lane; j < NPIX; j += 32) {
    if (j == wid) continue;
    float dx = (float)(xi - j % HH), dy = (float)(yi - j / HH);
    s += __expf(-INV_2SG2 * (dx * dx + dy * dy));
  }
#pragma unroll
  for (int o = 16; o >= 1; o >>= 1) s += __shfl_xor(s, o, 32);
  if (lane == 0) sg[wid] = s;
}

// row sums of bilateral kernel: one wave per (n,i) row
__global__ void sb_kernel(const float* __restrict__ colors, float* __restrict__ sb) {
  int wid  = (blockIdx.x * blockDim.x + threadIdx.x) >> 5;
  int lane = threadIdx.x & 31;
  if (wid >= NB * NPIX) return;
  int n = wid / NPIX, i = wid % NPIX;
  int yi = i / HH, xi = i % HH;
  const float* cb = colors + (size_t)n * NPIX * 3;
  float r0 = cb[(size_t)i * 3], g0 = cb[(size_t)i * 3 + 1], b0 = cb[(size_t)i * 3 + 2];
  float s = 0.f;
  for (int j = lane; j < NPIX; j += 32) {
    if (j == i) continue;
    float dx = (float)(xi - j % HH), dy = (float)(yi - j / HH);
    float dr = r0 - cb[(size_t)j * 3];
    float dg = g0 - cb[(size_t)j * 3 + 1];
    float db = b0 - cb[(size_t)j * 3 + 2];
    float cd2 = dr * dr + dg * dg + db * db;
    s += __expf(-INV_2SB2 * (dx * dx + dy * dy) - INV_2SR2 * cd2);
  }
#pragma unroll
  for (int o = 16; o >= 1; o >>= 1) s += __shfl_xor(s, o, 32);
  if (lane == 0) sb[wid] = s;
}

// Q0 (channel-major fp16, padded) into q0; zero q1 so padded lanes stay 0 in ping-pong
__global__ void initq_kernel(const float* __restrict__ probs,
                             _Float16* __restrict__ q0, _Float16* __restrict__ q1) {
  int idx = blockIdx.x * blockDim.x + threadIdx.x;
  if (idx >= NB * CP * NP) return;
  int n = idx / (CP * NP);
  int r = idx % (CP * NP);
  int c = r / NP, j = r % NP;
  _Float16 v = (_Float16)0.f;
  if (c < NC && j < NPIX) v = (_Float16)probs[((size_t)n * NC + c) * NPIX + j];
  q0[idx] = v;
  q1[idx] = (_Float16)0.f;
}

// combined, symmetric-normalized kernel Kc = 3*Kg_norm + 10*Kb_norm, fp16, row-major (n, NP, NP)
__global__ void kc_kernel(const float* __restrict__ colors, const float* __restrict__ sg,
                          const float* __restrict__ sb, _Float16* __restrict__ Kc) {
  size_t idx = (size_t)blockIdx.x * blockDim.x + threadIdx.x;
  if (idx >= (size_t)NB * NP * NP) return;
  int n = (int)(idx / ((size_t)NP * NP));
  int r = (int)(idx % ((size_t)NP * NP));
  int i = r / NP, j = r % NP;
  float v = 0.f;
  if (i < NPIX && j < NPIX && i != j) {
    float dx = (float)(i % HH - j % HH), dy = (float)(i / HH - j / HH);
    float d2 = dx * dx + dy * dy;
    float g = __expf(-INV_2SG2 * d2) * rsqrtf(sg[i] * sg[j]);
    const float* cb = colors + (size_t)n * NPIX * 3;
    float dr = cb[(size_t)i * 3]     - cb[(size_t)j * 3];
    float dg = cb[(size_t)i * 3 + 1] - cb[(size_t)j * 3 + 1];
    float db = cb[(size_t)i * 3 + 2] - cb[(size_t)j * 3 + 2];
    float cd2 = dr * dr + dg * dg + db * db;
    float b = __expf(-INV_2SB2 * d2 - INV_2SR2 * cd2)
              * rsqrtf(sb[(size_t)n * NPIX + i] * sb[(size_t)n * NPIX + j]);
    v = COMPAT_G * g + COMPAT_B * b;
  }
  Kc[idx] = (_Float16)v;
}

// one mean-field step: per wave a 16-pixel x 32-channel f32 tile via v_wmma_f32_16x16x32_f16,
// then channel softmax (+logu) through LDS; Qout written fp16 channel-major.
#define WPB 8   // 8 waves / 256 threads per block; 1696 jobs / 8 = 212 blocks exactly (no guard)
__global__ __launch_bounds__(256) void update_kernel(
    const _Float16* __restrict__ Kc, const _Float16* __restrict__ Qin,
    const float* __restrict__ logu, _Float16* __restrict__ Qout) {
  __shared__ float tile[WPB][16][33];
  int wave = threadIdx.x >> 5;
  int lane = threadIdx.x & 31;
  int job  = blockIdx.x * WPB + wave;         // exactly NB*(NP/16) jobs, no guard -> EXEC all 1s
  int n    = job / (NP / 16);
  int i0   = (job % (NP / 16)) * 16;

  const _Float16* A = Kc  + ((size_t)n * NP + i0) * NP;
  const _Float16* B = Qin + (size_t)n * CP * NP;

  int m  = lane & 15;
  int hk = (lane >> 4) << 4;                  // A half-lane K offset: 0 / 16
  int bk = (lane >> 4) << 4;                  // B half-lane K offset: 0 / 16
  const _Float16* arow = A + (size_t)m * NP + hk;
  const _Float16* b0p  = B + (size_t)m * NP + bk;          // channels 0..15
  const _Float16* b1p  = B + ((size_t)m + 16) * NP + bk;   // channels 16..31

  v8f acc0 = {}; v8f acc1 = {};
  for (int ks = 0; ks < KSTEPS; ++ks) {
    int k = ks << 5;
    // WGP-scope prefetch (locality 3): pull the streaming Kc line (L2-resident after
    // iteration 0) into WGP$ ahead of the clause of global_load_b128s.
    __builtin_prefetch((const void*)(arow + k + 64), 0, 3);
    v16h a  = *(const v16h*)(arow + k);         // K: hk+0..15
    v16h b0 = *(const v16h*)(b0p + k);          // K: bk+0..15 for column m
    v16h b1 = *(const v16h*)(b1p + k);
    acc0 = __builtin_amdgcn_wmma_f32_16x16x32_f16(false, a, false, b0, (short)0, acc0, false, false);
    acc1 = __builtin_amdgcn_wmma_f32_16x16x32_f16(false, a, false, b1, (short)0, acc1, false, false);
  }

  // D layout: lane<16 -> (pixel=v, chan=lane); lane>=16 -> (pixel=v+8, chan=lane-16)
  int prow = (lane >> 4) ? 8 : 0;
  int c0   = lane & 15;
#pragma unroll
  for (int v = 0; v < 8; ++v) {
    tile[wave][prow + v][c0]      = acc0[v];
    tile[wave][prow + v][c0 + 16] = acc1[v];
  }
  __syncthreads();

  if (lane < 16) {
    int i = i0 + lane;
    if (i < NPIX) {
      const float* lu = logu + (size_t)n * NC * NPIX + i;
      float lg[NC];
      float mx = -1e30f;
#pragma unroll
      for (int c = 0; c < NC; ++c) {
        lg[c] = lu[(size_t)c * NPIX] + tile[wave][lane][c];
        mx = fmaxf(mx, lg[c]);
      }
      float s = 0.f;
#pragma unroll
      for (int c = 0; c < NC; ++c) { lg[c] = __expf(lg[c] - mx); s += lg[c]; }
      float inv = 1.0f / s;
      _Float16* qo = Qout + (size_t)n * CP * NP + i;
#pragma unroll
      for (int c = 0; c < NC; ++c) qo[(size_t)c * NP] = (_Float16)(lg[c] * inv);
    }
  }
}

// final clamp/renorm + KL-ish loss, mean over (n,h,w)
__global__ void loss_kernel(const _Float16* __restrict__ Qfin,
                            const float* __restrict__ probs, float* __restrict__ out) {
  __shared__ float red[256];
  int idx = blockIdx.x * blockDim.x + threadIdx.x;
  float acc = 0.f;
  if (idx < NB * NPIX) {
    int n = idx / NPIX, i = idx % NPIX;
    const _Float16* q = Qfin + (size_t)n * CP * NP + i;
    float qs[NC]; float s = 0.f;
#pragma unroll
    for (int c = 0; c < NC; ++c) {
      float v = fmaxf((float)q[(size_t)c * NP], EPSV);
      qs[c] = v; s += v;
    }
    float inv = 1.0f / s;
    const float* p = probs + (size_t)n * NC * NPIX + i;
#pragma unroll
    for (int c = 0; c < NC; ++c) {
      float v = qs[c] * inv;
      float ratio = fminf(fmaxf(v / p[(size_t)c * NPIX], 0.05f), 20.0f);
      acc += v * __logf(ratio);
    }
  }
  red[threadIdx.x] = acc;
  __syncthreads();
#pragma unroll
  for (int o = 128; o >= 1; o >>= 1) {
    if (threadIdx.x < o) red[threadIdx.x] += red[threadIdx.x + o];
    __syncthreads();
  }
  if (threadIdx.x == 0) atomicAdd(out, red[0] * (1.0f / (NB * NPIX)));
}

extern "C" void kernel_launch(void* const* d_in, const int* in_sizes, int n_in,
                              void* d_out, int out_size, void* d_ws, size_t ws_size,
                              hipStream_t stream) {
  (void)in_sizes; (void)n_in; (void)out_size; (void)ws_size;
  const float* images  = (const float*)d_in[0];   // (16,3,321,321)
  const float* predict = (const float*)d_in[1];   // (16,21,41,41)
  float* out = (float*)d_out;

  char* ws = (char*)d_ws;
  size_t off = 0;
  auto take = [&](size_t bytes) -> char* {
    char* p = ws + off;
    off = (off + bytes + 255) & ~(size_t)255;
    return p;
  };
  float*    probs  = (float*)   take(sizeof(float) * (size_t)NB * NC * NPIX);
  float*    logu   = (float*)   take(sizeof(float) * (size_t)NB * NC * NPIX);
  float*    colors = (float*)   take(sizeof(float) * (size_t)NB * NPIX * 3);
  float*    sg     = (float*)   take(sizeof(float) * (size_t)NPIX);
  float*    sb     = (float*)   take(sizeof(float) * (size_t)NB * NPIX);
  _Float16* q0     = (_Float16*)take(sizeof(_Float16) * (size_t)NB * CP * NP);
  _Float16* q1     = (_Float16*)take(sizeof(_Float16) * (size_t)NB * CP * NP);
  _Float16* Kc     = (_Float16*)take(sizeof(_Float16) * (size_t)NB * NP * NP); // ~92MB, fits L2

  zero_kernel<<<1, 1, 0, stream>>>(out);

  int pixjobs = NB * NPIX;                               // 26896
  probs_kernel <<<(pixjobs + 255) / 256, 256, 0, stream>>>(predict, probs, logu);
  resize_kernel<<<(pixjobs + 255) / 256, 256, 0, stream>>>(images, colors);
  sg_kernel    <<<(NPIX + 7) / 8,        256, 0, stream>>>(sg);
  sb_kernel    <<<(pixjobs + 7) / 8,     256, 0, stream>>>(colors, sb);

  int qtot = NB * CP * NP;
  initq_kernel <<<(qtot + 255) / 256,    256, 0, stream>>>(probs, q0, q1);

  size_t kctot = (size_t)NB * NP * NP;
  kc_kernel    <<<(int)((kctot + 255) / 256), 256, 0, stream>>>(colors, sg, sb, Kc);

  _Float16* qa = q0; _Float16* qb = q1;
  for (int t = 0; t < 10; ++t) {
    update_kernel<<<(NB * (NP / 16)) / WPB, 256, 0, stream>>>(Kc, qa, logu, qb);
    _Float16* tmp = qa; qa = qb; qb = tmp;
  }

  loss_kernel<<<(pixjobs + 255) / 256, 256, 0, stream>>>(qa, probs, out);
}
